// VQVAEWithAttention_87428354278327
// MI455X (gfx1250) — matrix operand order repaired and checked by reading, not compiled, not run on hardware
//
#include <hip/hip_runtime.h>

typedef __attribute__((ext_vector_type(16))) _Float16 v16h;
typedef __attribute__((ext_vector_type(8)))  _Float16 v8h;
typedef __attribute__((ext_vector_type(4)))  _Float16 v4h;
typedef __attribute__((ext_vector_type(8)))  float    v8f;

#define B_SZ   16384
#define DIN    1280
#define DH     1024
#define DE     512
#define KCB    8192
#define BETA   0.25f

#define TILE_M 128
#define TILE_N 128
#define TILE_K 32
#define LDS_STRIDE 40   // 32 + 8 halves pad; 80B rows stay 16B-aligned

// ---------------------------------------------------------------- f32 -> f16
__global__ __launch_bounds__(256)
void cvt_f32_f16_kernel(const float* __restrict__ s, _Float16* __restrict__ d, int n4) {
    int i = blockIdx.x * blockDim.x + threadIdx.x;
    if (i < n4) {
        float4 v = ((const float4*)s)[i];
        v4h o = { (_Float16)v.x, (_Float16)v.y, (_Float16)v.z, (_Float16)v.w };
        ((v4h*)d)[i] = o;
    }
}

// ------------------------------------------------- WMMA GEMM: C = act(A·Wt+b)
// A: [M,K] f16 row-major (activations). W: [N,K] f16 row-major (torch layout).
// 128x128 block tile, 8 waves of 32x64 (2x4 WMMA), double-buffered LDS,
// one barrier per K-slice; next tile's global loads issued before the WMMAs.
__global__ __launch_bounds__(256)
void gemm_f16_wmma(const _Float16* __restrict__ A, const _Float16* __restrict__ W,
                   const float* __restrict__ bias, const float* __restrict__ residual,
                   float* __restrict__ outF32, _Float16* __restrict__ outF16,
                   int M, int N, int Kdim, int doRelu)
{
    __shared__ _Float16 As[2][TILE_M * LDS_STRIDE];
    __shared__ _Float16 Bs[2][TILE_N * LDS_STRIDE];

    const int tid     = threadIdx.x;
    const int lane    = tid & 31;
    const int wave    = tid >> 5;        // 0..7
    const int waveRow = wave >> 1;       // 0..3 -> 32-row slabs
    const int waveCol = wave & 1;        // 0..1 -> 64-col slabs
    const int blockM  = blockIdx.y * TILE_M;
    const int blockN  = blockIdx.x * TILE_N;
    const int half16  = lane >> 4;       // 0 | 1
    const int l15     = lane & 15;
    const int kbA     = half16 * 8;      // A frag: K 0-7/16-23 vs 8-15/24-31
    const int kbB     = half16 * 8;      // B frag: K 0-7/16-23 vs 8-15/24-31

    // per-thread staging slots: A vectors idx tid, tid+256; B vectors same
    const int rA0 = tid >> 2,          cgA0 = tid & 3;
    const int rA1 = (tid + 256) >> 2,  cgA1 = (tid + 256) & 3;

    v8f acc[2][4];
#pragma unroll
    for (int mi = 0; mi < 2; ++mi)
#pragma unroll
        for (int ni = 0; ni < 4; ++ni) acc[mi][ni] = (v8f){};

    v8h regA0, regA1, regB0, regB1;
    auto loadGlobal = [&](int k0) {
        regA0 = *(const v8h*)(A + (size_t)(blockM + rA0) * Kdim + k0 + cgA0 * 8);
        regA1 = *(const v8h*)(A + (size_t)(blockM + rA1) * Kdim + k0 + cgA1 * 8);
        regB0 = *(const v8h*)(W + (size_t)(blockN + rA0) * Kdim + k0 + cgA0 * 8);
        regB1 = *(const v8h*)(W + (size_t)(blockN + rA1) * Kdim + k0 + cgA1 * 8);
    };
    auto storeLDS = [&](int buf) {
        *(v8h*)(&As[buf][rA0 * LDS_STRIDE + cgA0 * 8]) = regA0;
        *(v8h*)(&As[buf][rA1 * LDS_STRIDE + cgA1 * 8]) = regA1;
        *(v8h*)(&Bs[buf][rA0 * LDS_STRIDE + cgA0 * 8]) = regB0;
        *(v8h*)(&Bs[buf][rA1 * LDS_STRIDE + cgA1 * 8]) = regB1;
    };

    loadGlobal(0);
    storeLDS(0);
    __syncthreads();

    int cur = 0;
    for (int k0 = 0; k0 < Kdim; k0 += TILE_K) {
        const bool hasNext = (k0 + TILE_K) < Kdim;   // uniform
        if (hasNext) loadGlobal(k0 + TILE_K);        // overlap with WMMAs below

        v16h afrag[2], bfrag[4];
#pragma unroll
        for (int mi = 0; mi < 2; ++mi) {
            int rowIn = waveRow * 32 + mi * 16 + l15;
            v8h lo = *(const v8h*)(&As[cur][rowIn * LDS_STRIDE + kbA]);
            v8h hi = *(const v8h*)(&As[cur][rowIn * LDS_STRIDE + kbA + 16]);
#pragma unroll
            for (int i = 0; i < 8; ++i) { afrag[mi][i] = lo[i]; afrag[mi][8 + i] = hi[i]; }
        }
#pragma unroll
        for (int ni = 0; ni < 4; ++ni) {
            int colIn = waveCol * 64 + ni * 16 + l15;
            v8h lo = *(const v8h*)(&Bs[cur][colIn * LDS_STRIDE + kbB]);
            v8h hi = *(const v8h*)(&Bs[cur][colIn * LDS_STRIDE + kbB + 16]);
#pragma unroll
            for (int i = 0; i < 8; ++i) { bfrag[ni][i] = lo[i]; bfrag[ni][8 + i] = hi[i]; }
        }
#pragma unroll
        for (int mi = 0; mi < 2; ++mi)
#pragma unroll
            for (int ni = 0; ni < 4; ++ni)
                acc[mi][ni] = __builtin_amdgcn_wmma_f32_16x16x32_f16(
                    false, afrag[mi], false, bfrag[ni], (short)0, acc[mi][ni], false, false);

        if (hasNext) storeLDS(cur ^ 1);
        __syncthreads();
        cur ^= 1;
    }

    // epilogue: D layout -> row = vgpr + 8*half16, col = l15
#pragma unroll
    for (int mi = 0; mi < 2; ++mi) {
#pragma unroll
        for (int ni = 0; ni < 4; ++ni) {
            int col = blockN + waveCol * 64 + ni * 16 + l15;
            float bv = bias ? bias[col] : 0.0f;
#pragma unroll
            for (int r = 0; r < 8; ++r) {
                int row = blockM + waveRow * 32 + mi * 16 + half16 * 8 + r;
                float v = acc[mi][ni][r] + bv;
                if (doRelu) v = v > 0.0f ? v : 0.0f;
                if (residual) v += residual[(size_t)row * N + col];
                if (outF32) outF32[(size_t)row * N + col] = v;
                if (outF16) outF16[(size_t)row * N + col] = (_Float16)v;
            }
        }
    }
}

// ---------------------------------------------------------------- LayerNorm
__global__ __launch_bounds__(256)
void layernorm_kernel(const float* __restrict__ x, const float* __restrict__ g,
                      const float* __restrict__ b, float* __restrict__ oF32,
                      _Float16* __restrict__ oF16, int D)
{
    __shared__ float red[256];
    const int row = blockIdx.x;
    const float* xr = x + (size_t)row * D;
    float s = 0.0f;
    for (int i = threadIdx.x; i < D; i += 256) s += xr[i];
    red[threadIdx.x] = s; __syncthreads();
    for (int o = 128; o > 0; o >>= 1) { if ((int)threadIdx.x < o) red[threadIdx.x] += red[threadIdx.x + o]; __syncthreads(); }
    float mean = red[0] / (float)D;
    __syncthreads();
    float vs = 0.0f;
    for (int i = threadIdx.x; i < D; i += 256) { float d = xr[i] - mean; vs += d * d; }
    red[threadIdx.x] = vs; __syncthreads();
    for (int o = 128; o > 0; o >>= 1) { if ((int)threadIdx.x < o) red[threadIdx.x] += red[threadIdx.x + o]; __syncthreads(); }
    float rstd = rsqrtf(red[0] / (float)D + 1e-5f);
    for (int i = threadIdx.x; i < D; i += 256) {
        float v = (xr[i] - mean) * rstd * g[i] + b[i];
        oF32[(size_t)row * D + i] = v;
        oF16[(size_t)row * D + i] = (_Float16)v;
    }
}

// -------------------------------------------------------- codebook ||c||^2
__global__ __launch_bounds__(128)
void cbnorm_kernel(const float* __restrict__ cb, float* __restrict__ cn)
{
    __shared__ float red[128];
    const int k = blockIdx.x;
    const float* c = cb + (size_t)k * DE;
    float s = 0.0f;
    for (int i = threadIdx.x; i < DE; i += 128) { float v = c[i]; s += v * v; }
    red[threadIdx.x] = s; __syncthreads();
    for (int o = 64; o > 0; o >>= 1) { if ((int)threadIdx.x < o) red[threadIdx.x] += red[threadIdx.x + o]; __syncthreads(); }
    if (threadIdx.x == 0) cn[k] = red[0];
}

// --------------------------- fused codebook search: WMMA scores + argmin
// One wave owns 16 rows of z; preloads all 16 A-frags (K=512); streams 8192
// codes in chunks of 16 (16 WMMAs/chunk). score = ||c||^2 - 2*(z.c).
__global__ __launch_bounds__(256)
void vq_argmin_kernel(const _Float16* __restrict__ z16, const _Float16* __restrict__ cb16,
                      const float* __restrict__ cn, int* __restrict__ idxOut)
{
    const int tid     = threadIdx.x;
    const int lane    = tid & 31;
    const int waveG   = blockIdx.x * 8 + (tid >> 5);
    const int rowBase = waveG * 16;
    const int l15     = lane & 15;
    const int half16  = lane >> 4;
    const int kbA     = half16 * 8;
    const int kbB     = half16 * 8;

    v16h afr[16];
#pragma unroll
    for (int kt = 0; kt < 16; ++kt) {
        const _Float16* zp = z16 + (size_t)(rowBase + l15) * DE + kt * 32;
        v8h lo = *(const v8h*)(zp + kbA);
        v8h hi = *(const v8h*)(zp + kbA + 16);
#pragma unroll
        for (int i = 0; i < 8; ++i) { afr[kt][i] = lo[i]; afr[kt][8 + i] = hi[i]; }
    }

    float bestV[8]; int bestI[8];
#pragma unroll
    for (int r = 0; r < 8; ++r) { bestV[r] = 3.0e38f; bestI[r] = 0; }

    for (int chunk = 0; chunk < KCB / 16; ++chunk) {
        const int col = chunk * 16 + l15;
        const _Float16* cp = cb16 + (size_t)col * DE;
        v8f acc = {};
#pragma unroll
        for (int kt = 0; kt < 16; ++kt) {
            v8h lo = *(const v8h*)(cp + kt * 32 + kbB);
            v8h hi = *(const v8h*)(cp + kt * 32 + kbB + 16);
            v16h bfr;
#pragma unroll
            for (int i = 0; i < 8; ++i) { bfr[i] = lo[i]; bfr[8 + i] = hi[i]; }
            acc = __builtin_amdgcn_wmma_f32_16x16x32_f16(
                false, afr[kt], false, bfr, (short)0, acc, false, false);
        }
        const float cnv = cn[col];
#pragma unroll
        for (int r = 0; r < 8; ++r) {
            float sc = cnv - 2.0f * acc[r];      // row = half16*8 + r, this lane's col
            if (sc < bestV[r]) { bestV[r] = sc; bestI[r] = col; }
        }
    }

    // min-reduce across the 16 lanes that share each row group (stay in halves)
#pragma unroll
    for (int m = 8; m >= 1; m >>= 1) {
#pragma unroll
        for (int r = 0; r < 8; ++r) {
            float ov = __shfl_xor(bestV[r], m, 32);
            int   oi = __shfl_xor(bestI[r], m, 32);
            if (ov < bestV[r] || (ov == bestV[r] && oi < bestI[r])) { bestV[r] = ov; bestI[r] = oi; }
        }
    }
    if (l15 == 0) {
#pragma unroll
        for (int r = 0; r < 8; ++r)
            idxOut[rowBase + half16 * 8 + r] = bestI[r];
    }
}

// ----------------------- gather quant rows (f16) + per-row ||q - z||^2 (f32)
__global__ __launch_bounds__(256)
void vq_gather_loss_kernel(const float* __restrict__ z, const float* __restrict__ cb,
                           const int* __restrict__ idx, _Float16* __restrict__ quant16,
                           float* __restrict__ rowloss)
{
    __shared__ float red[256];
    const int row = blockIdx.x;
    const int k = idx[row];
    const float* c  = cb + (size_t)k * DE;
    const float* zr = z + (size_t)row * DE;
    float s = 0.0f;
    for (int i = threadIdx.x; i < DE; i += 256) {
        float cv = c[i];
        quant16[(size_t)row * DE + i] = (_Float16)cv;
        float d = cv - zr[i];
        s += d * d;
    }
    red[threadIdx.x] = s; __syncthreads();
    for (int o = 128; o > 0; o >>= 1) { if ((int)threadIdx.x < o) red[threadIdx.x] += red[threadIdx.x + o]; __syncthreads(); }
    if (threadIdx.x == 0) rowloss[row] = red[0];
}

// deterministic fixed-order final reduction (single block)
__global__ __launch_bounds__(256)
void loss_reduce_kernel(const float* __restrict__ rowloss, float* __restrict__ lossOut)
{
    __shared__ float red[256];
    float s = 0.0f;
    for (int i = threadIdx.x; i < B_SZ; i += 256) s += rowloss[i];
    red[threadIdx.x] = s; __syncthreads();
    for (int o = 128; o > 0; o >>= 1) { if ((int)threadIdx.x < o) red[threadIdx.x] += red[threadIdx.x + o]; __syncthreads(); }
    if (threadIdx.x == 0)
        lossOut[0] = red[0] * ((1.0f + BETA) / ((float)B_SZ * (float)DE));
}

// ============================================================== launch
extern "C" void kernel_launch(void* const* d_in, const int* in_sizes, int n_in,
                              void* d_out, int out_size, void* d_ws, size_t ws_size,
                              hipStream_t stream) {
    (void)in_sizes; (void)n_in; (void)out_size; (void)ws_size;
    const float* x    = (const float*)d_in[0];
    const float* w_in = (const float*)d_in[1];
    const float* b_in = (const float*)d_in[2];
    const float* w_v  = (const float*)d_in[3];
    const float* b_v  = (const float*)d_in[4];
    const float* w_o  = (const float*)d_in[5];
    const float* b_o  = (const float*)d_in[6];
    const float* ln_g = (const float*)d_in[7];
    const float* ln_b = (const float*)d_in[8];
    const float* w1   = (const float*)d_in[9];
    const float* b1   = (const float*)d_in[10];
    const float* w2   = (const float*)d_in[11];
    const float* b2   = (const float*)d_in[12];
    const float* w_cb = (const float*)d_in[13];
    const float* b_cb = (const float*)d_in[14];
    const float* cb   = (const float*)d_in[15];
    const float* w_d1 = (const float*)d_in[16];
    const float* b_d1 = (const float*)d_in[17];
    const float* w_d2 = (const float*)d_in[18];
    const float* b_d2 = (const float*)d_in[19];
    float* out = (float*)d_out;

    char* ws = (char*)d_ws;
    size_t off = 0;
    auto alloc = [&](size_t bytes) { size_t o = off; off += (bytes + 255) & ~(size_t)255; return o; };

    _Float16* w_in16 = (_Float16*)(ws + alloc((size_t)DH * DIN * 2));
    _Float16* w_v16  = (_Float16*)(ws + alloc((size_t)DH * DH * 2));
    _Float16* w_o16  = (_Float16*)(ws + alloc((size_t)DH * DH * 2));
    _Float16* w1_16  = (_Float16*)(ws + alloc((size_t)DH * DH * 2));
    _Float16* w2_16  = (_Float16*)(ws + alloc((size_t)DH * DH * 2));
    _Float16* wcb16  = (_Float16*)(ws + alloc((size_t)DE * DH * 2));
    _Float16* cb16   = (_Float16*)(ws + alloc((size_t)KCB * DE * 2));
    _Float16* wd1_16 = (_Float16*)(ws + alloc((size_t)DH * DE * 2));
    _Float16* wd2_16 = (_Float16*)(ws + alloc((size_t)DIN * DH * 2));
    // lifetime-aliased activation buffers
    _Float16* A16  = (_Float16*)(ws + alloc((size_t)B_SZ * DIN * 2)); // x16 -> d1_16
    _Float16* B16  = (_Float16*)(ws + alloc((size_t)B_SZ * DH * 2));  // h16 -> hn16 -> h2_16 -> quant16
    _Float16* C16  = (_Float16*)(ws + alloc((size_t)B_SZ * DH * 2));  // t16 -> m1 -> z16
    float*    F32A = (float*)(ws + alloc((size_t)B_SZ * DH * 4));     // h -> hn
    float*    F32B = (float*)(ws + alloc((size_t)B_SZ * DH * 4));     // attn+res -> z
    float*    cn   = (float*)(ws + alloc((size_t)KCB * 4));
    int*      idx  = (int*)(ws + alloc((size_t)B_SZ * 4));
    float*    rlos = (float*)(ws + alloc((size_t)B_SZ * 4));

    auto cvt = [&](const float* s, _Float16* d, size_t n) {
        int n4 = (int)(n / 4);
        cvt_f32_f16_kernel<<<dim3((n4 + 255) / 256), dim3(256), 0, stream>>>(s, d, n4);
    };
    auto gemm = [&](const _Float16* A, const _Float16* W, const float* bias, const float* res,
                    float* o32, _Float16* o16, int M, int N, int K, int relu) {
        gemm_f16_wmma<<<dim3(N / TILE_N, M / TILE_M), dim3(256), 0, stream>>>(
            A, W, bias, res, o32, o16, M, N, K, relu);
    };

    // f16 conversions
    cvt(x,    A16,    (size_t)B_SZ * DIN);
    cvt(w_in, w_in16, (size_t)DH * DIN);
    cvt(w_v,  w_v16,  (size_t)DH * DH);
    cvt(w_o,  w_o16,  (size_t)DH * DH);
    cvt(w1,   w1_16,  (size_t)DH * DH);
    cvt(w2,   w2_16,  (size_t)DH * DH);
    cvt(w_cb, wcb16,  (size_t)DE * DH);
    cvt(cb,   cb16,   (size_t)KCB * DE);
    cvt(w_d1, wd1_16, (size_t)DH * DE);
    cvt(w_d2, wd2_16, (size_t)DIN * DH);

    // encoder
    gemm(A16, w_in16, b_in, nullptr, F32A, B16, B_SZ, DH, DIN, 0);   // h (f32+f16)
    gemm(B16, w_v16,  b_v,  nullptr, nullptr, C16, B_SZ, DH, DH, 0); // v = h@Wv
    gemm(C16, w_o16,  b_o,  F32A,   F32B, nullptr, B_SZ, DH, DH, 0); // attn + h
    layernorm_kernel<<<dim3(B_SZ), dim3(256), 0, stream>>>(F32B, ln_g, ln_b, F32A, B16, DH); // hn
    gemm(B16, w1_16,  b1,   nullptr, nullptr, C16, B_SZ, DH, DH, 1); // m1 = relu(.)
    gemm(C16, w2_16,  b2,   F32A,   nullptr, B16, B_SZ, DH, DH, 1);  // h2 = relu(.)+hn
    gemm(B16, wcb16,  b_cb, nullptr, F32B,   C16, B_SZ, DE, DH, 0);  // z (f32+f16)

    // vector quantization
    cbnorm_kernel<<<dim3(KCB), dim3(128), 0, stream>>>(cb, cn);
    vq_argmin_kernel<<<dim3(B_SZ / 16 / 8), dim3(256), 0, stream>>>(C16, cb16, cn, idx);
    vq_gather_loss_kernel<<<dim3(B_SZ), dim3(256), 0, stream>>>(F32B, cb, idx, B16, rlos);
    loss_reduce_kernel<<<dim3(1), dim3(256), 0, stream>>>(rlos, out + (size_t)B_SZ * DIN);

    // decoder
    gemm(B16, wd1_16, b_d1, nullptr, nullptr, A16, B_SZ, DH, DE, 1); // d1 = relu(.)
    gemm(A16, wd2_16, b_d2, nullptr, out,  nullptr, B_SZ, DIN, DH, 1); // out = relu(.)
}
